// MultiheadAttention_KV_1726576853358
// MI455X (gfx1250) — matrix operand order repaired and checked by reading, not compiled, not run on hardware
//
#include <hip/hip_runtime.h>
#include <hip/hip_bf16.h>
#include <math.h>

typedef __attribute__((ext_vector_type(16))) _Float16 v16h;
typedef __attribute__((ext_vector_type(8)))  _Float16 v8h;
typedef __attribute__((ext_vector_type(8)))  float    v8f;

#define B_   8
#define S_   1024
#define IN_  1024
#define EMB_ 1024
#define H_   16
#define HD_  64
#define PD_  10

// ---------------------------------------------------------------------------
// WMMA helpers (CDNA5 gfx1250, wave32)
// ---------------------------------------------------------------------------
__device__ __forceinline__ v8f wmma_f16(v16h a, v16h b, v8f c) {
    // D = A(16x32 f16) x B(32x16 f16) + C(16x16 f32)
    return __builtin_amdgcn_wmma_f32_16x16x32_f16(
        false, a, false, b, (short)0, c, false, false);
}

// A operand: 16x32 tile, row-major source [.., ld]. Lane owns row (lane&15).
// halves 0..15 = K (hi?16:0)+0..15
__device__ __forceinline__ v16h load_a_f16(const _Float16* __restrict__ p,
                                           int ld, int mBase, int kBase) {
    const int lane = threadIdx.x & 31;
    const int hi   = lane >> 4;
    const int row  = mBase + (lane & 15);
    const _Float16* q = p + row * ld + kBase + hi * 16;
    v8h lo = *(const v8h*)q;
    v8h hh = *(const v8h*)(q + 8);
    return __builtin_shufflevector(lo, hh, 0,1,2,3,4,5,6,7,8,9,10,11,12,13,14,15);
}

// B operand: 32x16 tile; column n = source row (nBase + lane&15),
// K = kBase + hi*16 + 0..15, contiguous 32B per lane.
__device__ __forceinline__ v16h load_b_f16(const _Float16* __restrict__ p,
                                           int ld, int nBase, int kBase) {
    const int lane = threadIdx.x & 31;
    const int hi   = lane >> 4;
    const int row  = nBase + (lane & 15);
    return *(const v16h*)(p + row * ld + kBase + hi * 16);
}

// ---------------------------------------------------------------------------
// Shared GEMM core: one wave computes a 32(M) x 64(N) tile, K-loop software
// pipelined with double-buffered operand registers + prefetch one chunk ahead.
// ---------------------------------------------------------------------------
__device__ __forceinline__ void gemm_wave_32x64(const _Float16* __restrict__ A,
                                                const _Float16* __restrict__ Bsrc,
                                                int ld, int K,
                                                int mBase, int nBase,
                                                v8f acc[8]) {
    v16h a0 = load_a_f16(A, ld, mBase,      0);
    v16h a1 = load_a_f16(A, ld, mBase + 16, 0);
    v16h b0 = load_b_f16(Bsrc, ld, nBase,      0);
    v16h b1 = load_b_f16(Bsrc, ld, nBase + 16, 0);
    v16h b2 = load_b_f16(Bsrc, ld, nBase + 32, 0);
    v16h b3 = load_b_f16(Bsrc, ld, nBase + 48, 0);

    const int lane = threadIdx.x & 31;
    for (int kk = 32; kk <= K; kk += 32) {
        v16h a0n = a0, a1n = a1, b0n = b0, b1n = b1, b2n = b2, b3n = b3;
        if (kk < K) {
            a0n = load_a_f16(A, ld, mBase,      kk);
            a1n = load_a_f16(A, ld, mBase + 16, kk);
            b0n = load_b_f16(Bsrc, ld, nBase,      kk);
            b1n = load_b_f16(Bsrc, ld, nBase + 16, kk);
            b2n = load_b_f16(Bsrc, ld, nBase + 32, kk);
            b3n = load_b_f16(Bsrc, ld, nBase + 48, kk);
            if (kk + 32 < K) {  // prefetch one chunk further ahead
                __builtin_prefetch(A    + (mBase + (lane & 15)) * ld + kk + 32, 0, 1);
                __builtin_prefetch(Bsrc + (nBase + (lane & 15)) * ld + kk + 32, 0, 1);
            }
        }
        acc[0] = wmma_f16(a0, b0, acc[0]);
        acc[1] = wmma_f16(a0, b1, acc[1]);
        acc[2] = wmma_f16(a0, b2, acc[2]);
        acc[3] = wmma_f16(a0, b3, acc[3]);
        acc[4] = wmma_f16(a1, b0, acc[4]);
        acc[5] = wmma_f16(a1, b1, acc[5]);
        acc[6] = wmma_f16(a1, b2, acc[6]);
        acc[7] = wmma_f16(a1, b3, acc[7]);
        a0 = a0n; a1 = a1n; b0 = b0n; b1 = b1n; b2 = b2n; b3 = b3n;
    }
}

// ---------------------------------------------------------------------------
// Elementwise f32 -> f16 conversion, 8 elements/thread (b128 traffic)
// ---------------------------------------------------------------------------
__global__ __launch_bounds__(256) void cvt_f16_kernel(const float* __restrict__ in,
                                                      _Float16* __restrict__ out, int n) {
    int i = (blockIdx.x * 256 + threadIdx.x) * 8;
    if (i < n) {
        float4 f0 = *(const float4*)(in + i);
        float4 f1 = *(const float4*)(in + i + 4);
        v8h o;
        o[0] = (_Float16)f0.x; o[1] = (_Float16)f0.y;
        o[2] = (_Float16)f0.z; o[3] = (_Float16)f0.w;
        o[4] = (_Float16)f1.x; o[5] = (_Float16)f1.y;
        o[6] = (_Float16)f1.z; o[7] = (_Float16)f1.w;
        *(v8h*)(out + i) = o;
    }
}

// ---------------------------------------------------------------------------
// pos_bias[q,k] = dot(pos[0,q,k,:], mp_w) + mp_b   (S*S elements)
// ---------------------------------------------------------------------------
__global__ __launch_bounds__(256) void posbias_kernel(const float* __restrict__ pos,
                                                      const float* __restrict__ mpw,
                                                      const float* __restrict__ mpb,
                                                      float* __restrict__ pb) {
    int i = blockIdx.x * 256 + threadIdx.x;
    if (i < S_ * S_) {
        const float* p = pos + (long)i * PD_;
        float s = 0.f;
#pragma unroll
        for (int j = 0; j < PD_; ++j) s += p[j] * mpw[j];
        pb[i] = s + mpb[0];
    }
}

// ---------------------------------------------------------------------------
// KV GEMM:  kv[m, c] = sum_k xh[m,k]*wh[c,k] + kv_b[c]
//   Scatter: k -> kh [B,H,S,HD]  (c%128 < 64)
//            v -> vT [B,H,HD,S]  (c%128 >= 64, key-contiguous for PV GEMM)
// Block: 256 thr = 8 waves. Wave w: rows mBase..+31, cols blockIdx.y*512+w*64
// ---------------------------------------------------------------------------
__global__ __launch_bounds__(256) void kv_gemm_kernel(const _Float16* __restrict__ xh,
                                                      const _Float16* __restrict__ wh,
                                                      const float* __restrict__ kvb,
                                                      _Float16* __restrict__ kh,
                                                      _Float16* __restrict__ vT) {
    const int wave  = threadIdx.x >> 5;
    const int lane  = threadIdx.x & 31;
    const int laneN = lane & 15;
    const int hi    = lane >> 4;
    const int mBase = blockIdx.x * 32;
    const int nBase = blockIdx.y * 512 + wave * 64;

    v8f acc[8] = {};
    gemm_wave_32x64(xh, wh, IN_, IN_, mBase, nBase, acc);

#pragma unroll
    for (int mt = 0; mt < 2; ++mt) {
#pragma unroll
        for (int t = 0; t < 4; ++t) {
            const v8f a = acc[mt * 4 + t];
#pragma unroll
            for (int r = 0; r < 8; ++r) {
                const int m = mBase + mt * 16 + r + hi * 8;
                const int c = nBase + t * 16 + laneN;
                const float v = a[r] + kvb[c];
                const int b    = m >> 10;       // m / S
                const int s    = m & (S_ - 1);  // m % S
                const int head = c >> 7;        // c / (2*HD)
                const int w    = c & 127;       // c % (2*HD)
                if (w < HD_)
                    kh[(((b * H_ + head) * S_) + s) * HD_ + w] = (_Float16)v;
                else
                    vT[(((b * H_ + head) * HD_) + (w - HD_)) * S_ + s] = (_Float16)v;
            }
        }
    }
}

// ---------------------------------------------------------------------------
// Flash attention: 4 waves/block, each wave owns one 16-query tile of (b,h).
//   scores = (q_k . k) * (sum(mp_w)/sqrt(HD)) + pos_bias[q,k]
//   online softmax in WMMA C layout, O += P x V, normalize, write vals f16.
// Key-block loop is pipelined: next K-tiles + current V-tiles are issued
// before the softmax VALU section so loads overlap exp/shuffle work.
// ---------------------------------------------------------------------------
__global__ __launch_bounds__(128) void attn_kernel(const _Float16* __restrict__ kh,
                                                   const _Float16* __restrict__ vT,
                                                   const float* __restrict__ posb,
                                                   const float* __restrict__ mpw,
                                                   _Float16* __restrict__ valsh) {
    const int wave  = threadIdx.x >> 5;
    const int lane  = threadIdx.x & 31;
    const int laneN = lane & 15;
    const int hi    = lane >> 4;

    const int qTiles = S_ / 64;                 // 64 queries per block
    const int bid = blockIdx.x;
    const int b   = bid / (H_ * qTiles);
    const int rem = bid % (H_ * qTiles);
    const int h   = rem / qTiles;
    const int qb  = (rem % qTiles) * 64 + wave * 16;

    const _Float16* Kp = kh + (long)(b * H_ + h) * S_ * HD_;  // [S, HD]
    const _Float16* Vp = vT + (long)(b * H_ + h) * HD_ * S_;  // [HD, S]

    float wsum = 0.f;
#pragma unroll
    for (int p = 0; p < PD_; ++p) wsum += mpw[p];
    const float scale = wsum * 0.125f;  // 1/sqrt(HD) = 1/8

    const v16h aq0 = load_a_f16(Kp, HD_, qb, 0);
    const v16h aq1 = load_a_f16(Kp, HD_, qb, 32);

    v8f o0 = {}, o1 = {}, o2 = {}, o3 = {};
    float mrow[8], lrow[8];
#pragma unroll
    for (int r = 0; r < 8; ++r) { mrow[r] = -INFINITY; lrow[r] = 0.f; }

    __shared__ _Float16 pbuf_all[4 * 16 * 32];  // per-wave P staging (1KB each)
    _Float16* pbuf = pbuf_all + wave * (16 * 32);

    // preload first key block's K operands
    v16h bk0 = load_b_f16(Kp, HD_, 0,  0);
    v16h bk1 = load_b_f16(Kp, HD_, 0,  32);
    v16h bk2 = load_b_f16(Kp, HD_, 16, 0);
    v16h bk3 = load_b_f16(Kp, HD_, 16, 32);

    for (int kb = 0; kb < S_; kb += 32) {
        // ---- scores: two 16x16 C tiles over 32 keys, K-dim = 64 ----
        v8f s0 = {}, s1 = {};
        s0 = wmma_f16(aq0, bk0, s0);
        s0 = wmma_f16(aq1, bk1, s0);
        s1 = wmma_f16(aq0, bk2, s1);
        s1 = wmma_f16(aq1, bk3, s1);

        // ---- issue V tiles for this block + K tiles for next block now,
        //      so they overlap the softmax VALU section ----
        v16h bv0 = load_b_f16(Vp, S_, 0,  kb);
        v16h bv1 = load_b_f16(Vp, S_, 16, kb);
        v16h bv2 = load_b_f16(Vp, S_, 32, kb);
        v16h bv3 = load_b_f16(Vp, S_, 48, kb);
        if (kb + 32 < S_) {
            bk0 = load_b_f16(Kp, HD_, kb + 32, 0);
            bk1 = load_b_f16(Kp, HD_, kb + 32, 32);
            bk2 = load_b_f16(Kp, HD_, kb + 48, 0);
            bk3 = load_b_f16(Kp, HD_, kb + 48, 32);
        }

        // ---- online softmax over this 32-key slab ----
#pragma unroll
        for (int r = 0; r < 8; ++r) {
            const int q = qb + r + hi * 8;
            float x0 = s0[r] * scale + posb[q * S_ + kb + laneN];
            float x1 = s1[r] * scale + posb[q * S_ + kb + 16 + laneN];

            float rmax = fmaxf(x0, x1);
#pragma unroll
            for (int msk = 1; msk < 16; msk <<= 1)
                rmax = fmaxf(rmax, __shfl_xor(rmax, msk, 32));

            const float mold = mrow[r];
            const float mnew = fmaxf(mold, rmax);
            const float p0 = __expf(x0 - mnew);
            const float p1 = __expf(x1 - mnew);
            float rs = p0 + p1;
#pragma unroll
            for (int msk = 1; msk < 16; msk <<= 1)
                rs += __shfl_xor(rs, msk, 32);

            const float corr = __expf(mold - mnew);  // exp(-inf)=0 first pass
            lrow[r] = lrow[r] * corr + rs;
            mrow[r] = mnew;
            o0[r] *= corr; o1[r] *= corr; o2[r] *= corr; o3[r] *= corr;

            const int row = r + hi * 8;
            pbuf[row * 32 + laneN]      = (_Float16)p0;
            pbuf[row * 32 + 16 + laneN] = (_Float16)p1;
        }
        __syncthreads();
        const v16h pa = load_a_f16(pbuf, 32, 0, 0);  // 16 q x 32 keys (A layout)
        __syncthreads();

        // ---- O += P x V : 4 dim-tiles of 16 ----
        o0 = wmma_f16(pa, bv0, o0);
        o1 = wmma_f16(pa, bv1, o1);
        o2 = wmma_f16(pa, bv2, o2);
        o3 = wmma_f16(pa, bv3, o3);
    }

    // ---- normalize + write vals [B,S,EMB] f16 ----
#pragma unroll
    for (int r = 0; r < 8; ++r) {
        const float inv = 1.0f / lrow[r];
        const int q = qb + r + hi * 8;
        _Float16* dst = valsh + (long)(b * S_ + q) * EMB_ + h * HD_;
        dst[0  + laneN] = (_Float16)(o0[r] * inv);
        dst[16 + laneN] = (_Float16)(o1[r] * inv);
        dst[32 + laneN] = (_Float16)(o2[r] * inv);
        dst[48 + laneN] = (_Float16)(o3[r] * inv);
    }
}

// ---------------------------------------------------------------------------
// Output GEMM: out[m, c] = sum_k valsh[m,k]*owh[c,k] + o_b[c]   (f32 out)
// ---------------------------------------------------------------------------
__global__ __launch_bounds__(256) void out_gemm_kernel(const _Float16* __restrict__ vh,
                                                       const _Float16* __restrict__ owh,
                                                       const float* __restrict__ ob,
                                                       float* __restrict__ out) {
    const int wave  = threadIdx.x >> 5;
    const int lane  = threadIdx.x & 31;
    const int laneN = lane & 15;
    const int hi    = lane >> 4;
    const int mBase = blockIdx.x * 32;
    const int nBase = blockIdx.y * 512 + wave * 64;

    v8f acc[8] = {};
    gemm_wave_32x64(vh, owh, EMB_, EMB_, mBase, nBase, acc);

#pragma unroll
    for (int mt = 0; mt < 2; ++mt) {
#pragma unroll
        for (int t = 0; t < 4; ++t) {
            const v8f a = acc[mt * 4 + t];
#pragma unroll
            for (int r = 0; r < 8; ++r) {
                const int m = mBase + mt * 16 + r + hi * 8;
                const int c = nBase + t * 16 + laneN;
                out[(long)m * EMB_ + c] = a[r] + ob[c];
            }
        }
    }
}

// ---------------------------------------------------------------------------
extern "C" void kernel_launch(void* const* d_in, const int* in_sizes, int n_in,
                              void* d_out, int out_size, void* d_ws, size_t ws_size,
                              hipStream_t stream) {
    const float* x    = (const float*)d_in[0];
    const float* pos  = (const float*)d_in[1];
    const float* kv_w = (const float*)d_in[2];
    const float* kv_b = (const float*)d_in[3];
    const float* o_w  = (const float*)d_in[4];
    const float* o_b  = (const float*)d_in[5];
    const float* mp_w = (const float*)d_in[6];
    const float* mp_b = (const float*)d_in[7];
    float* out = (float*)d_out;

    char* ws = (char*)d_ws;
    const size_t MB = 1024ull * 1024ull;
    _Float16* xh    = (_Float16*)(ws + 0 * MB);   // 16 MB : x f16       [B*S, IN]
    _Float16* wh    = (_Float16*)(ws + 16 * MB);  //  4 MB : kv_w f16    [2EMB, IN]
    _Float16* owh   = (_Float16*)(ws + 20 * MB);  //  2 MB : o_w f16     [EMB, EMB]
    _Float16* kh    = (_Float16*)(ws + 22 * MB);  // 16 MB : k  [B,H,S,HD]
    _Float16* vT    = (_Float16*)(ws + 38 * MB);  // 16 MB : vT [B,H,HD,S]
    float*    posb  = (float*)   (ws + 54 * MB);  //  4 MB : pos_bias [S,S]
    _Float16* valsh = (_Float16*)(ws + 58 * MB);  // 16 MB : vals f16 [B*S, EMB]

    const int n_x  = B_ * S_ * IN_;      // 8388608
    const int n_w  = 2 * EMB_ * IN_;     // 2097152
    const int n_ow = EMB_ * EMB_;        // 1048576

    cvt_f16_kernel<<<(n_x  / 8 + 255) / 256, 256, 0, stream>>>(x,    xh,  n_x);
    cvt_f16_kernel<<<(n_w  / 8 + 255) / 256, 256, 0, stream>>>(kv_w, wh,  n_w);
    cvt_f16_kernel<<<(n_ow / 8 + 255) / 256, 256, 0, stream>>>(o_w,  owh, n_ow);

    posbias_kernel<<<(S_ * S_ + 255) / 256, 256, 0, stream>>>(pos, mp_w, mp_b, posb);

    // KV projection: M = B*S = 8192 (256 tiles of 32), N = 2048 (4 x 512)
    kv_gemm_kernel<<<dim3(256, 4), 256, 0, stream>>>(xh, wh, kv_b, kh, vT);

    // Flash attention: B*H*(S/64) = 2048 blocks x 4 waves
    attn_kernel<<<B_ * H_ * (S_ / 64), 128, 0, stream>>>(kh, vT, posb, mp_w, valsh);

    // Output projection: M = 8192 (256 tiles of 32), N = 1024 (2 x 512)
    out_gemm_kernel<<<dim3(256, 2), 256, 0, stream>>>(valsh, owh, o_b, out);
}
